// attention_single_83932250898677
// MI455X (gfx1250) — compile-verified, matrix operand order checked by reading, not run on hardware
//
#include <hip/hip_runtime.h>
#include <hip/hip_bf16.h>

// Problem constants (fixed by the reference)
#define BATCH 4
#define SEQ   2048
#define DIM   1024
#define LN_EPS 1e-5f
#define SCORE_SCALE (1.0f / 32.0f)   // BETA / sqrt(D) = 1/sqrt(1024)

typedef __bf16 v16bf __attribute__((ext_vector_type(16)));
typedef float  v8f   __attribute__((ext_vector_type(8)));

__device__ __forceinline__ __bf16 f2bf(float f) {
  union { float f; unsigned u; } in; in.f = f;
  unsigned r = in.u + 0x7FFFu + ((in.u >> 16) & 1u);   // RNE
  union { unsigned short s; __bf16 b; } out;
  out.s = (unsigned short)(r >> 16);
  return out.b;
}

// ---------------------------------------------------------------------------
// Fragment gather for wave32 16x16x32 bf16 WMMA.
// Per ISA 7.12.2: lanes 0-15 hold row (lane), K = {kb..kb+7, kb+16..kb+23};
// lanes 16-31 hold row (lane-16), K = {kb+8..kb+15, kb+24..kb+31}.
// Same pattern for A (row = M) and B (row = N of Bt). Two 16B loads per lane.
// ---------------------------------------------------------------------------
__device__ __forceinline__ v16bf load_frag(const __bf16* __restrict__ p,
                                           int ld, int row0, int kb) {
  const int lane = threadIdx.x & 31;
  const __bf16* q = p + (size_t)(row0 + (lane & 15)) * ld + kb + ((lane >> 4) << 3);
  union { uint4 u[2]; v16bf v; } t;
  t.u[0] = *(const uint4*)(q);
  t.u[1] = *(const uint4*)(q + 16);
  return t.v;
}

// ---------------------------------------------------------------------------
// Generic "A * Bt^T" GEMM:  C[m,n] = scale * sum_k A[m,k]*Bt[n,k] + bias[n]
// A  : bf16 row-major [M, Kd], leading dim lda, per-batch stride sA (elems)
// Bt : bf16 row-major [N, Kd], leading dim ldb, per-batch stride sB (elems)
// MODE 0: store f32 row-major      C[m*ldc + n]
// MODE 1: store bf16 row-major     C[m*ldc + n]
// MODE 2: store bf16 transposed    C[n*ldc + m]   (ldc = M leading dim)
// Block = 128 threads = 4 waves; block tile 128(M) x 64(N); wave tile 32x64.
// K-loop is register double-buffered (Kd must be a multiple of 64).
// ---------------------------------------------------------------------------
template <int MODE>
__global__ __launch_bounds__(128)
void gemm_bt_wmma(const __bf16* __restrict__ A, long long sA, int lda,
                  const __bf16* __restrict__ Bt, long long sB, int ldb,
                  void* __restrict__ Cv, long long sC, int ldc,
                  const float* __restrict__ bias, float scale, int Kd) {
  const int wave = threadIdx.x >> 5;
  const int lane = threadIdx.x & 31;
  const int n0 = blockIdx.x * 64;
  const int m0 = blockIdx.y * 128 + wave * 32;
  const int b  = blockIdx.z;

  const __bf16* Ab = A  + (size_t)b * (size_t)sA;
  const __bf16* Bb = Bt + (size_t)b * (size_t)sB;

  v8f acc[2][4] = {};
  v16bf a0[2], b0[4], a1[2], b1[4];

  // Prologue: fragments for kb = 0
#pragma unroll
  for (int i = 0; i < 2; ++i) a0[i] = load_frag(Ab, lda, m0 + 16 * i, 0);
#pragma unroll
  for (int j = 0; j < 4; ++j) b0[j] = load_frag(Bb, ldb, n0 + 16 * j, 0);

  for (int kb = 0; kb < Kd; kb += 64) {
    // Issue loads for kb+32 before consuming kb's fragments.
#pragma unroll
    for (int i = 0; i < 2; ++i) a1[i] = load_frag(Ab, lda, m0 + 16 * i, kb + 32);
#pragma unroll
    for (int j = 0; j < 4; ++j) b1[j] = load_frag(Bb, ldb, n0 + 16 * j, kb + 32);

    if (kb + 128 < Kd) {  // distant speculative prefetch -> global_prefetch_b8
      __builtin_prefetch((const void*)(Ab + (size_t)(m0 + (lane & 15)) * lda + kb + 128), 0, 1);
    }

#pragma unroll
    for (int i = 0; i < 2; ++i)
#pragma unroll
      for (int j = 0; j < 4; ++j)
        acc[i][j] = __builtin_amdgcn_wmma_f32_16x16x32_bf16(
            false, a0[i], false, b0[j], (short)0, acc[i][j], false, false);

    if (kb + 64 < Kd) {  // issue loads for kb+64 before consuming kb+32's
#pragma unroll
      for (int i = 0; i < 2; ++i) a0[i] = load_frag(Ab, lda, m0 + 16 * i, kb + 64);
#pragma unroll
      for (int j = 0; j < 4; ++j) b0[j] = load_frag(Bb, ldb, n0 + 16 * j, kb + 64);
    }

#pragma unroll
    for (int i = 0; i < 2; ++i)
#pragma unroll
      for (int j = 0; j < 4; ++j)
        acc[i][j] = __builtin_amdgcn_wmma_f32_16x16x32_bf16(
            false, a1[i], false, b1[j], (short)0, acc[i][j], false, false);
  }

  // C/D layout (ISA 7.12.2): vgpr r, lane l -> M = r + 8*(l>=16), N = l%16
  const int fn   = lane & 15;
  const int mofs = (lane >> 4) << 3;
#pragma unroll
  for (int j = 0; j < 4; ++j) {
    const int gn = n0 + 16 * j + fn;
    const float bv = bias ? bias[gn] : 0.0f;
#pragma unroll
    for (int i = 0; i < 2; ++i) {
#pragma unroll
      for (int r = 0; r < 8; ++r) {
        const int gm = m0 + 16 * i + mofs + r;
        const float v = acc[i][j][r] * scale + bv;
        if (MODE == 0) {
          float* C = (float*)Cv + (size_t)b * (size_t)sC;
          C[(size_t)gm * ldc + gn] = v;
        } else if (MODE == 1) {
          __bf16* C = (__bf16*)Cv + (size_t)b * (size_t)sC;
          C[(size_t)gm * ldc + gn] = f2bf(v);
        } else {
          __bf16* C = (__bf16*)Cv + (size_t)b * (size_t)sC;
          C[(size_t)gn * ldc + gm] = f2bf(v);   // transposed store (for V^T)
        }
      }
    }
  }
}

// ---------------------------------------------------------------------------
// fp32 -> bf16 bulk convert
// ---------------------------------------------------------------------------
__global__ __launch_bounds__(256)
void cvt_bf16_kernel(const float* __restrict__ in, __bf16* __restrict__ out, int n) {
  int i = blockIdx.x * 256 + threadIdx.x;
  if (i < n) out[i] = f2bf(in[i]);
}

// W [D,D] fp32 row-major  ->  Wt [D,D] bf16 with Wt[n,d] = W[d,n]
__global__ __launch_bounds__(256)
void transpose_w_bf16_kernel(const float* __restrict__ W, __bf16* __restrict__ Wt) {
  int idx = blockIdx.x * 256 + threadIdx.x;      // DIM*DIM total
  int d = idx & (DIM - 1);
  int n = idx >> 10;                              // DIM = 1024
  Wt[idx] = f2bf(W[(size_t)d * DIM + n]);
}

// ---------------------------------------------------------------------------
// Row softmax over S=2048: in-place fp32 normalize + bf16 copy for next GEMM.
// One block (256 thr) per row, 8 elements per thread.
// ---------------------------------------------------------------------------
__global__ __launch_bounds__(256)
void softmax_row_kernel(float* __restrict__ Wgt, __bf16* __restrict__ Wb) {
  const int row = blockIdx.x;
  float* p = Wgt + (size_t)row * SEQ;
  __bf16* pb = Wb + (size_t)row * SEQ;
  __shared__ float red[256];
  const int t = threadIdx.x;

  float v[8];
  float mx = -1e30f;
#pragma unroll
  for (int i = 0; i < 8; ++i) {
    v[i] = p[t + 256 * i];
    mx = fmaxf(mx, v[i]);
  }
  red[t] = mx; __syncthreads();
  for (int s = 128; s > 0; s >>= 1) {
    if (t < s) red[t] = fmaxf(red[t], red[t + s]);
    __syncthreads();
  }
  mx = red[0]; __syncthreads();

  float sum = 0.0f;
#pragma unroll
  for (int i = 0; i < 8; ++i) {
    v[i] = __expf(v[i] - mx);
    sum += v[i];
  }
  red[t] = sum; __syncthreads();
  for (int s = 128; s > 0; s >>= 1) {
    if (t < s) red[t] += red[t + s];
    __syncthreads();
  }
  const float inv = 1.0f / red[0];

#pragma unroll
  for (int i = 0; i < 8; ++i) {
    const float w = v[i] * inv;
    p[t + 256 * i] = w;
    pb[t + 256 * i] = f2bf(w);
  }
}

// ---------------------------------------------------------------------------
// out = x + att ; LayerNorm over D=1024 -> out1.  One block per row.
// ---------------------------------------------------------------------------
__global__ __launch_bounds__(256)
void residual_ln_kernel(const float* __restrict__ x, const float* __restrict__ att,
                        const float* __restrict__ gamma, const float* __restrict__ beta,
                        float* __restrict__ out1) {
  const int row = blockIdx.x;
  const float* px = x + (size_t)row * DIM;
  const float* pa = att + (size_t)row * DIM;
  float* po = out1 + (size_t)row * DIM;
  __shared__ float red[256];
  const int t = threadIdx.x;

  float v[4];
  float s = 0.0f, s2 = 0.0f;
#pragma unroll
  for (int i = 0; i < 4; ++i) {
    const int c = t + 256 * i;
    v[i] = px[c] + pa[c];
    s += v[i];
    s2 += v[i] * v[i];
  }
  red[t] = s; __syncthreads();
  for (int k = 128; k > 0; k >>= 1) { if (t < k) red[t] += red[t + k]; __syncthreads(); }
  const float mu = red[0] * (1.0f / DIM); __syncthreads();
  red[t] = s2; __syncthreads();
  for (int k = 128; k > 0; k >>= 1) { if (t < k) red[t] += red[t + k]; __syncthreads(); }
  const float var = red[0] * (1.0f / DIM) - mu * mu;
  const float inv = rsqrtf(var + LN_EPS);

#pragma unroll
  for (int i = 0; i < 4; ++i) {
    const int c = t + 256 * i;
    po[c] = (v[i] - mu) * inv * gamma[c] + beta[c];
  }
}

// ---------------------------------------------------------------------------
extern "C" void kernel_launch(void* const* d_in, const int* in_sizes, int n_in,
                              void* d_out, int out_size, void* d_ws, size_t ws_size,
                              hipStream_t stream) {
  const float* x     = (const float*)d_in[0];
  const float* Wk    = (const float*)d_in[1];
  const float* bk    = (const float*)d_in[2];
  const float* Wq    = (const float*)d_in[3];
  const float* bq    = (const float*)d_in[4];
  const float* Wv    = (const float*)d_in[5];
  const float* bv    = (const float*)d_in[6];
  const float* gamma = (const float*)d_in[7];
  const float* betaL = (const float*)d_in[8];

  const size_t BSD = (size_t)BATCH * SEQ * DIM;     // 8,388,608
  const size_t BSS = (size_t)BATCH * SEQ * SEQ;     // 16,777,216

  float* out1   = (float*)d_out;                    // [B,S,D]
  float* att    = out1 + BSD;                       // [B,S,D]
  float* weight = att + BSD;                        // [B,S,S] (raw scores -> softmax in place)

  // Workspace carve-out (all bf16), 256B aligned
  char* ws = (char*)d_ws;
  size_t off = 0;
  auto carve = [&](size_t bytes) -> void* {
    void* p = ws + off;
    off = (off + bytes + 255) & ~(size_t)255;
    return p;
  };
  __bf16* xb  = (__bf16*)carve(BSD * 2);
  __bf16* Wtk = (__bf16*)carve((size_t)DIM * DIM * 2);
  __bf16* Wtq = (__bf16*)carve((size_t)DIM * DIM * 2);
  __bf16* Wtv = (__bf16*)carve((size_t)DIM * DIM * 2);
  __bf16* Qb  = (__bf16*)carve(BSD * 2);
  __bf16* Kb  = (__bf16*)carve(BSD * 2);
  __bf16* Vt  = (__bf16*)carve(BSD * 2);            // per-batch [D, S] (transposed V)
  __bf16* Wb  = (__bf16*)carve(BSS * 2);            // softmax weights bf16

  // --- prep: convert activations and transpose weight matrices to bf16 ---
  cvt_bf16_kernel<<<(int)((BSD + 255) / 256), 256, 0, stream>>>(x, xb, (int)BSD);
  const int wdd = (DIM * DIM) / 256;
  transpose_w_bf16_kernel<<<wdd, 256, 0, stream>>>(Wk, Wtk);
  transpose_w_bf16_kernel<<<wdd, 256, 0, stream>>>(Wq, Wtq);
  transpose_w_bf16_kernel<<<wdd, 256, 0, stream>>>(Wv, Wtv);

  const long long SD = (long long)SEQ * DIM;
  const long long SS = (long long)SEQ * SEQ;
  const long long DS = (long long)DIM * SEQ;

  // --- QKV projections: [S,D] = xb[b] @ Wt^T + bias, per batch ---
  dim3 gproj(DIM / 64, SEQ / 128, BATCH);
  gemm_bt_wmma<1><<<gproj, 128, 0, stream>>>(xb, SD, DIM, Wtq, 0, DIM,
                                             Qb, SD, DIM, bq, 1.0f, DIM);
  gemm_bt_wmma<1><<<gproj, 128, 0, stream>>>(xb, SD, DIM, Wtk, 0, DIM,
                                             Kb, SD, DIM, bk, 1.0f, DIM);
  gemm_bt_wmma<2><<<gproj, 128, 0, stream>>>(xb, SD, DIM, Wtv, 0, DIM,
                                             Vt, DS, SEQ, bv, 1.0f, DIM);  // V^T store

  // --- scores = scale * Q @ K^T  (natively A*Bt^T form) -> weight (f32) ---
  dim3 gsc(SEQ / 64, SEQ / 128, BATCH);
  gemm_bt_wmma<0><<<gsc, 128, 0, stream>>>(Qb, SD, DIM, Kb, SD, DIM,
                                           weight, SS, SEQ, nullptr, SCORE_SCALE, DIM);

  // --- row softmax (fp32 output in place + bf16 copy for next GEMM) ---
  softmax_row_kernel<<<BATCH * SEQ, 256, 0, stream>>>(weight, Wb);

  // --- att = weight @ V  == A(Wb) * Bt(V^T)^T -> att (f32) ---
  dim3 gatt(DIM / 64, SEQ / 128, BATCH);
  gemm_bt_wmma<0><<<gatt, 128, 0, stream>>>(Wb, SS, SEQ, Vt, DS, SEQ,
                                            att, SD, DIM, nullptr, 1.0f, SEQ);

  // --- residual + LayerNorm -> out1 ---
  residual_ln_kernel<<<BATCH * SEQ, 256, 0, stream>>>(x, att, gamma, betaL, out1);
}
